// YuanAttention_5317169512814
// MI455X (gfx1250) — matrix operand order repaired and checked by reading, not compiled, not run on hardware
//
#include <hip/hip_runtime.h>

// Problem constants (match reference)
#define TT   4096   // B*S tokens
#define HH   2048   // hidden
#define SS   1024   // seq len
#define BBB  4      // batch
#define NHH  16     // heads
#define HDD  128    // head dim
#define PP   2048   // NH*HD

typedef __attribute__((ext_vector_type(16))) __bf16 v16bf;
typedef __attribute__((ext_vector_type(8)))  __bf16 v8bf;
typedef __attribute__((ext_vector_type(8)))  float  v8f;

static __device__ inline v8f vzero8() {
  v8f z;
#pragma unroll
  for (int i = 0; i < 8; ++i) z[i] = 0.f;
  return z;
}

static __device__ inline v16bf mk16(v8bf a, v8bf b) {
  return __builtin_shufflevector(a, b, 0,1,2,3,4,5,6,7,8,9,10,11,12,13,14,15);
}

static __device__ inline v8f wmma_bf16(v16bf a, v16bf b, v8f c) {
  return __builtin_amdgcn_wmma_f32_16x16x32_bf16(false, a, false, b, (short)0, c, false, false);
}

// Async global->LDS copy, 16B per lane.  VDST = per-lane LDS byte address,
// VADDR = per-lane 64-bit global address.  Tracked by ASYNCcnt.
static __device__ inline void async_ld16(unsigned lds_off, const void* gptr) {
  asm volatile("global_load_async_to_lds_b128 %0, %1, off"
               :: "v"(lds_off), "v"(gptr) : "memory");
}

template <int N>
static __device__ inline void wait_async() {
  asm volatile("s_wait_asynccnt %0" :: "n"(N) : "memory");
}

// XOR swizzle of 16B chunks within a 64B (4-chunk) LDS row.
static __device__ inline int sx4(int r) { return (r & 3) ^ ((r >> 2) & 3); }

// ---------------------------------------------------------------------------
// bf16 WMMA GEMM with async-LDS double buffering:
//   C[M,N] = A[M,K] @ B0[N,K]^T (+bias)
// CONV adds the fused 2-tap sequence conv:
//   C[t] = A_prev[t] @ B0^T + A[t] @ B1^T + bias,
//   A_prev[t] = (t%S==0) ? cacheA[t/S] : A[t-1]
// Tiles 128x128x32; 256 threads (8 waves); wave tile 64x32 (4x2 WMMA accs).
// LDS tiles are unpadded 64B-pitch with XOR chunk swizzle.
// ---------------------------------------------------------------------------
template <bool CONV, bool BIAS, bool OUTBF>
__global__ __launch_bounds__(256)
void gemm_async(void* __restrict__ Cv,
                const __bf16* __restrict__ A,
                const __bf16* __restrict__ B0,
                const __bf16* __restrict__ B1,
                const float* __restrict__ bias,
                const __bf16* __restrict__ cacheA,
                int M, int N, int K) {
  constexpr int CSZ = CONV ? 128 * 32 : 1;
  __shared__ __bf16 sA0[2][128 * 32];
  __shared__ __bf16 sB0[2][128 * 32];
  __shared__ __bf16 sA1[2][CSZ];
  __shared__ __bf16 sB1[2][CSZ];

  const int tid  = threadIdx.x;
  const int m0   = blockIdx.y * 128;
  const int n0   = blockIdx.x * 128;
  const int wv   = tid >> 5;
  const int lane = tid & 31;
  const int lh   = lane >> 4;
  const int lm   = lane & 15;
  const int wm   = (wv >> 2) * 64;
  const int wn   = (wv & 3) * 32;

  // Two 16B chunks per wave per 8KB tile (8 waves x 2 x 32 lanes x 16B).
  const int ch0 = (wv * 2 + 0) * 32 + lane;
  const int ch1 = (wv * 2 + 1) * 32 + lane;

  auto issue = [&](int k0, int buf) {
    const int chs[2] = {ch0, ch1};
#pragma unroll
    for (int j = 0; j < 2; ++j) {
      const int r  = chs[j] >> 2;          // tile row (4 chunks per 64B row)
      const int c  = chs[j] & 3;           // chunk in row
      const int cs = c ^ sx4(r);           // swizzled chunk
      const unsigned eoff = (unsigned)(r * 32 + cs * 8);
      if constexpr (CONV) {
        const int gr = m0 + r;
        const __bf16* rp = ((gr & (SS - 1)) == 0)
                               ? (cacheA + (size_t)(gr >> 10) * K)
                               : (A + (size_t)(gr - 1) * K);
        async_ld16((unsigned)(size_t)&sA0[buf][eoff], rp + k0 + c * 8);
        async_ld16((unsigned)(size_t)&sA1[buf][eoff],
                   A + (size_t)gr * K + k0 + c * 8);
        async_ld16((unsigned)(size_t)&sB0[buf][eoff],
                   B0 + (size_t)(n0 + r) * K + k0 + c * 8);
        async_ld16((unsigned)(size_t)&sB1[buf][eoff],
                   B1 + (size_t)(n0 + r) * K + k0 + c * 8);
      } else {
        async_ld16((unsigned)(size_t)&sA0[buf][eoff],
                   A + (size_t)(m0 + r) * K + k0 + c * 8);
        async_ld16((unsigned)(size_t)&sB0[buf][eoff],
                   B0 + (size_t)(n0 + r) * K + k0 + c * 8);
      }
    }
  };

  v8f acc[4][2];
#pragma unroll
  for (int i = 0; i < 4; ++i)
#pragma unroll
    for (int j = 0; j < 2; ++j) acc[i][j] = vzero8();

  const int niter = K >> 5;
  issue(0, 0);

  for (int it = 0; it < niter; ++it) {
    const int buf = it & 1;
    if (it + 1 < niter) {
      issue((it + 1) << 5, buf ^ 1);
      wait_async<CONV ? 8 : 4>();   // current tile landed; next stays in flight
    } else {
      wait_async<0>();
    }
    __syncthreads();                 // everyone's tile landed

    v16bf af[4], bfr[2];
#pragma unroll
    for (int i = 0; i < 4; ++i) {
      const int row = wm + i * 16 + lm;
      const int s   = sx4(row);
      af[i] = mk16(*(const v8bf*)&sA0[buf][row * 32 + (((2 * lh) ^ s) << 3)],
                   *(const v8bf*)&sA0[buf][row * 32 + (((2 * lh + 1) ^ s) << 3)]);
    }
#pragma unroll
    for (int j = 0; j < 2; ++j) {
      const int nn = wn + j * 16 + lm;
      const int s  = sx4(nn);
      bfr[j] = mk16(*(const v8bf*)&sB0[buf][nn * 32 + (((2 * lh) ^ s) << 3)],
                    *(const v8bf*)&sB0[buf][nn * 32 + (((2 * lh + 1) ^ s) << 3)]);
    }
#pragma unroll
    for (int i = 0; i < 4; ++i)
#pragma unroll
      for (int j = 0; j < 2; ++j)
        acc[i][j] = wmma_bf16(af[i], bfr[j], acc[i][j]);

    if constexpr (CONV) {
      v16bf af1[4], bfr1[2];
#pragma unroll
      for (int i = 0; i < 4; ++i) {
        const int row = wm + i * 16 + lm;
        const int s   = sx4(row);
        af1[i] = mk16(*(const v8bf*)&sA1[buf][row * 32 + (((2 * lh) ^ s) << 3)],
                      *(const v8bf*)&sA1[buf][row * 32 + (((2 * lh + 1) ^ s) << 3)]);
      }
#pragma unroll
      for (int j = 0; j < 2; ++j) {
        const int nn = wn + j * 16 + lm;
        const int s  = sx4(nn);
        bfr1[j] = mk16(*(const v8bf*)&sB1[buf][nn * 32 + (((2 * lh) ^ s) << 3)],
                       *(const v8bf*)&sB1[buf][nn * 32 + (((2 * lh + 1) ^ s) << 3)]);
      }
#pragma unroll
      for (int i = 0; i < 4; ++i)
#pragma unroll
        for (int j = 0; j < 2; ++j)
          acc[i][j] = wmma_bf16(af1[i], bfr1[j], acc[i][j]);
    }
    __syncthreads();                 // reads done: next issue may overwrite
  }

  // Epilogue. C/D layout: lane holds col=lm, rows lh*8+r.
#pragma unroll
  for (int i = 0; i < 4; ++i)
#pragma unroll
    for (int j = 0; j < 2; ++j) {
      const int col = n0 + wn + j * 16 + lm;
      float bv = 0.f;
      if constexpr (BIAS) bv = bias[col];
#pragma unroll
      for (int r = 0; r < 8; ++r) {
        const int row = m0 + wm + i * 16 + lh * 8 + r;
        const float v = acc[i][j][r] + bv;
        if constexpr (OUTBF)
          ((__bf16*)Cv)[(size_t)row * N + col] = (__bf16)v;
        else
          ((float*)Cv)[(size_t)row * N + col] = v;
      }
    }
}

// ---------------------------------------------------------------------------
// f32 -> bf16 conversion, optional element stride/offset (conv tap split).
// ---------------------------------------------------------------------------
__global__ __launch_bounds__(256)
void cvt_bf16(__bf16* __restrict__ dst, const float* __restrict__ src,
              int n, int stride, int off) {
  int i = blockIdx.x * 256 + threadIdx.x;
  if (i < n) dst[i] = (__bf16)src[(size_t)i * stride + off];
}

// ---------------------------------------------------------------------------
// Residual + RMSNorm: LFb(bf16) = rmsnorm(O2 + X) * g.  Block per row.
// ---------------------------------------------------------------------------
__global__ __launch_bounds__(256)
void rmsnorm_res(__bf16* __restrict__ LFb, const float* __restrict__ O2,
                 const float* __restrict__ X, const float* __restrict__ g) {
  const int row = blockIdx.x, tid = threadIdx.x;
  const size_t base = (size_t)row * HH;
  float v[8];
  float ss = 0.f;
#pragma unroll
  for (int i = 0; i < 8; ++i) {
    int c = tid + i * 256;
    float x = O2[base + c] + X[base + c];
    v[i] = x;
    ss += x * x;
  }
#pragma unroll
  for (int off = 16; off; off >>= 1) ss += __shfl_xor(ss, off, 32);
  __shared__ float sb[8];
  if ((tid & 31) == 0) sb[tid >> 5] = ss;
  __syncthreads();
  float tot = 0.f;
#pragma unroll
  for (int i = 0; i < 8; ++i) tot += sb[i];
  const float inv = rsqrtf(tot * (1.0f / HH) + 1e-6f);
#pragma unroll
  for (int i = 0; i < 8; ++i) {
    int c = tid + i * 256;
    LFb[base + c] = (__bf16)(v[i] * inv * g[c]);
  }
}

// ---------------------------------------------------------------------------
// In-place RoPE on bf16 QK [T, NH, 2, HD] (first 64 dims rotated).
// ---------------------------------------------------------------------------
__global__ __launch_bounds__(256)
void rope_bf(__bf16* __restrict__ QKb, const float* __restrict__ rot) {
  int idx = blockIdx.x * 256 + threadIdx.x;   // T*NH*2*32
  int i  = idx & 31;
  int t2 = idx >> 5;
  int qk = t2 & 1;
  int hh = (t2 >> 1) & 15;
  int t  = t2 >> 5;
  int s  = t & (SS - 1);
  size_t base = (size_t)t * (2 * PP) + (size_t)hh * (2 * HDD) + qk * HDD;
  float f = rot[s * 64 + i];                  // freqs[i] == freqs[i+32]
  float c = __cosf(f), sn = __sinf(f);
  float a  = (float)QKb[base + i];
  float b2 = (float)QKb[base + i + 32];
  QKb[base + i]      = (__bf16)(a * c  - b2 * sn);
  QKb[base + i + 32] = (__bf16)(b2 * c + a  * sn);
}

// ---------------------------------------------------------------------------
// Causal flash attention, bf16 operands.  Block = 128 threads (4 waves)
// handles (b, h, 64 q rows).  K tile staged via async-LDS copies.
// ---------------------------------------------------------------------------
__global__ __launch_bounds__(128)
void flash_attn(__bf16* __restrict__ AOb, const __bf16* __restrict__ Qk,
                const __bf16* __restrict__ Vb) {
  const int qb = blockIdx.x, h = blockIdx.y, b = blockIdx.z;
  const int tid  = threadIdx.x;
  const int w    = tid >> 5, lane = tid & 31;
  const int lh   = lane >> 4, lm = lane & 15;

  __shared__ __bf16 Kb[32 * 128];    // [key][dim], 256B rows, chunk-swizzled
  __shared__ __bf16 Vt[128 * 32];    // [dim][key], 64B rows, chunk-swizzled
  __shared__ __bf16 Ps[4 * 16 * 32]; // per-wave P staging

  // Preload 16 q rows as 4 A-fragments directly from global bf16.
  const int   qrow  = qb * 64 + w * 16 + lm;
  const __bf16* qp  = Qk + ((size_t)(b * SS + qrow)) * (2 * PP) +
                      (size_t)h * (2 * HDD);
  v16bf qf[4];
#pragma unroll
  for (int s4 = 0; s4 < 4; ++s4)
    qf[s4] = mk16(*(const v8bf*)(qp + s4 * 32 + lh * 16),
                  *(const v8bf*)(qp + s4 * 32 + lh * 16 + 8));

  float mrow[8], lrow[8];
  v8f o[8];
#pragma unroll
  for (int r = 0; r < 8; ++r) { mrow[r] = -3.0e38f; lrow[r] = 0.f; }
#pragma unroll
  for (int j = 0; j < 8; ++j) o[j] = vzero8();

  const int nkb = 2 * (qb + 1);
  for (int kb = 0; kb < nkb; ++kb) {
    __syncthreads();                 // prior LDS reads complete
    // K tile (32 keys x 128 dims): 512 16B chunks, 4 async instr per wave.
#pragma unroll
    for (int j = 0; j < 4; ++j) {
      const int id = (w * 4 + j) * 32 + lane;
      const int r = id >> 4, c = id & 15;                 // 16 chunks per row
      const int cs = c ^ ((r & 3) << 2);
      async_ld16((unsigned)(size_t)&Kb[r * 128 + cs * 8],
                 Qk + ((size_t)(b * SS + kb * 32 + r)) * (2 * PP) +
                     (size_t)h * (2 * HDD) + HDD + c * 8);
    }
    // V tile transposed manually: Vt[dim][key].
#pragma unroll
    for (int itv = 0; itv < 4; ++itv) {
      const int id = tid + itv * 128;
      const int r = id >> 4, c = id & 15;                 // key r, dim chunk c
      const v8bf vv = *(const v8bf*)(Vb +
          ((size_t)(b * SS + kb * 32 + r)) * PP + (size_t)h * HDD + c * 8);
#pragma unroll
      for (int e = 0; e < 8; ++e) {
        const int dim = c * 8 + e;
        const int cs  = (r >> 3) ^ sx4(dim);
        Vt[dim * 32 + cs * 8 + (r & 7)] = vv[e];
      }
    }
    wait_async<0>();
    __syncthreads();                 // K + V staged for all waves

    // Scores: 16 q rows x 32 keys over HD=128.
    v8f sc[2];
    sc[0] = vzero8(); sc[1] = vzero8();
#pragma unroll
    for (int s4 = 0; s4 < 4; ++s4) {
#pragma unroll
      for (int j = 0; j < 2; ++j) {
        const int key = j * 16 + lm;
        const int kx  = (key & 3) << 2;
        const int c0  = (s4 * 4 + lh * 2) ^ kx;
        const int c1  = (s4 * 4 + lh * 2 + 1) ^ kx;
        v16bf bfr = mk16(*(const v8bf*)&Kb[key * 128 + c0 * 8],
                         *(const v8bf*)&Kb[key * 128 + c1 * 8]);
        sc[j] = wmma_bf16(qf[s4], bfr, sc[j]);
      }
    }

    const float SCL = 0.08838834764831845f;   // HD^-0.5
#pragma unroll
    for (int j = 0; j < 2; ++j)
#pragma unroll
      for (int r = 0; r < 8; ++r) {
        const int col = kb * 32 + j * 16 + lm;
        const int row = qb * 64 + w * 16 + lh * 8 + r;
        const float val = sc[j][r] * SCL;
        sc[j][r] = (col <= row) ? val : -3.0e38f;
      }

    // Online softmax (row reductions within each 16-lane half-wave).
    float mn[8];
#pragma unroll
    for (int r = 0; r < 8; ++r) mn[r] = fmaxf(sc[0][r], sc[1][r]);
#pragma unroll
    for (int off = 8; off; off >>= 1)
#pragma unroll
      for (int r = 0; r < 8; ++r) mn[r] = fmaxf(mn[r], __shfl_xor(mn[r], off, 32));
    float al[8];
#pragma unroll
    for (int r = 0; r < 8; ++r) {
      const float mnew = fmaxf(mrow[r], mn[r]);
      al[r] = __expf(mrow[r] - mnew);
      mrow[r] = mnew;
    }
#pragma unroll
    for (int j = 0; j < 2; ++j)
#pragma unroll
      for (int r = 0; r < 8; ++r) sc[j][r] = __expf(sc[j][r] - mrow[r]);
    float ps[8];
#pragma unroll
    for (int r = 0; r < 8; ++r) ps[r] = sc[0][r] + sc[1][r];
#pragma unroll
    for (int off = 8; off; off >>= 1)
#pragma unroll
      for (int r = 0; r < 8; ++r) ps[r] += __shfl_xor(ps[r], off, 32);
#pragma unroll
    for (int r = 0; r < 8; ++r) lrow[r] = lrow[r] * al[r] + ps[r];
#pragma unroll
    for (int j2 = 0; j2 < 8; ++j2)
#pragma unroll
      for (int r = 0; r < 8; ++r) o[j2][r] *= al[r];

    // P: C-layout -> A-layout through per-wave LDS (in-order DS ops).
#pragma unroll
    for (int j = 0; j < 2; ++j)
#pragma unroll
      for (int r = 0; r < 8; ++r)
        Ps[(w * 16 + lh * 8 + r) * 32 + j * 16 + lm] = (__bf16)sc[j][r];
    v16bf pf = mk16(*(const v8bf*)&Ps[(w * 16 + lm) * 32 + lh * 16],
                    *(const v8bf*)&Ps[(w * 16 + lm) * 32 + lh * 16 + 8]);

    // PV: 16x128 output, 8 WMMAs (K = 32 keys).
#pragma unroll
    for (int j2 = 0; j2 < 8; ++j2) {
      const int dim = j2 * 16 + lm;
      const int s   = sx4(dim);
      v16bf vfr = mk16(*(const v8bf*)&Vt[dim * 32 + (((lh * 2) ^ s) << 3)],
                       *(const v8bf*)&Vt[dim * 32 + (((lh * 2 + 1) ^ s) << 3)]);
      o[j2] = wmma_bf16(pf, vfr, o[j2]);
    }
  }

  // Normalize and write bf16 [T, P].
#pragma unroll
  for (int j2 = 0; j2 < 8; ++j2)
#pragma unroll
    for (int r = 0; r < 8; ++r) {
      const int row = qb * 64 + w * 16 + lh * 8 + r;
      const int col = h * HDD + j2 * 16 + lm;
      AOb[((size_t)(b * SS + row)) * PP + col] = (__bf16)(o[j2][r] / lrow[r]);
    }
}

// ---------------------------------------------------------------------------
extern "C" void kernel_launch(void* const* d_in, const int* in_sizes, int n_in,
                              void* d_out, int out_size, void* d_ws, size_t ws_size,
                              hipStream_t stream) {
  (void)in_sizes; (void)n_in; (void)out_size; (void)ws_size;
  const float* X   = (const float*)d_in[1];
  const float* rot = (const float*)d_in[2];
  const float* lf1 = (const float*)d_in[3];
  const float* lf2 = (const float*)d_in[4];
  const float* Wqk = (const float*)d_in[5];
  const float* Wv  = (const float*)d_in[6];
  const float* Wo  = (const float*)d_in[7];
  const float* c1w = (const float*)d_in[8];
  const float* c1b = (const float*)d_in[9];
  const float* c2w = (const float*)d_in[10];
  const float* c2b = (const float*)d_in[11];
  const float* lng = (const float*)d_in[12];
  float* out = (float*)d_out;

  // Workspace carve-out (256B aligned).
  char* cur = (char*)d_ws;
  auto alloc = [&](size_t bytes) {
    void* r = (void*)cur;
    cur += (bytes + 255) & ~(size_t)255;
    return r;
  };
  __bf16* Xb   = (__bf16*)alloc((size_t)TT * HH * 2);
  __bf16* Wvb  = (__bf16*)alloc((size_t)PP * HH * 2);
  __bf16* Wqkb = (__bf16*)alloc((size_t)2 * PP * HH * 2);
  __bf16* Wob  = (__bf16*)alloc((size_t)HH * PP * 2);
  __bf16* W1t0 = (__bf16*)alloc((size_t)(HH / 2) * HH * 2);
  __bf16* W1t1 = (__bf16*)alloc((size_t)(HH / 2) * HH * 2);
  __bf16* W2t0 = (__bf16*)alloc((size_t)HH * (HH / 2) * 2);
  __bf16* W2t1 = (__bf16*)alloc((size_t)HH * (HH / 2) * 2);
  __bf16* lf1b = (__bf16*)alloc((size_t)BBB * HH * 2);
  __bf16* lf2b = (__bf16*)alloc((size_t)BBB * (HH / 2) * 2);
  __bf16* Vb16 = (__bf16*)alloc((size_t)TT * PP * 2);
  __bf16* O1b  = (__bf16*)alloc((size_t)TT * (HH / 2) * 2);
  float*  O2f  = (float*) alloc((size_t)TT * HH * 4);
  __bf16* LFb  = (__bf16*)alloc((size_t)TT * HH * 2);
  __bf16* QKb  = (__bf16*)alloc((size_t)TT * 2 * PP * 2);
  __bf16* AOb  = (__bf16*)alloc((size_t)TT * PP * 2);

  dim3 blk(256);
  auto cvt = [&](__bf16* d, const float* s, int n, int st, int of) {
    cvt_bf16<<<dim3((n + 255) / 256), blk, 0, stream>>>(d, s, n, st, of);
  };
  // One-time bf16 conversions (weights + activations).
  cvt(Xb,   X,   TT * HH, 1, 0);
  cvt(Wvb,  Wv,  PP * HH, 1, 0);
  cvt(Wqkb, Wqk, 2 * PP * HH, 1, 0);
  cvt(Wob,  Wo,  HH * PP, 1, 0);
  cvt(W1t0, c1w, (HH / 2) * HH, 2, 0);
  cvt(W1t1, c1w, (HH / 2) * HH, 2, 1);
  cvt(W2t0, c2w, HH * (HH / 2), 2, 0);
  cvt(W2t1, c2w, HH * (HH / 2), 2, 1);
  cvt(lf1b, lf1, BBB * HH, 1, 0);
  cvt(lf2b, lf2, BBB * (HH / 2), 1, 0);

  // 1) V = X @ Wv^T                      (bf16 out, consumed by attention)
  gemm_async<false, false, true><<<dim3(PP / 128, TT / 128), blk, 0, stream>>>(
      Vb16, Xb, Wvb, nullptr, nullptr, nullptr, TT, PP, HH);
  // 2) O1 = conv1(X, lf1_cache)          (fused two-tap, bf16 out)
  gemm_async<true, true, true><<<dim3((HH / 2) / 128, TT / 128), blk, 0, stream>>>(
      O1b, Xb, W1t0, W1t1, c1b, lf1b, TT, HH / 2, HH);
  // 3) O2 = conv2(O1, lf2_cache)         (f32 out for RMSNorm accuracy)
  gemm_async<true, true, false><<<dim3(HH / 128, TT / 128), blk, 0, stream>>>(
      O2f, O1b, W2t0, W2t1, c2b, lf2b, TT, HH, HH / 2);
  // 4) LF = rmsnorm(O2 + X) * g          (bf16 out)
  rmsnorm_res<<<dim3(TT), blk, 0, stream>>>(LFb, O2f, X, lng);
  // 5) QK = LF @ Wqk^T                   (bf16 out)
  gemm_async<false, false, true><<<dim3((2 * PP) / 128, TT / 128), blk, 0, stream>>>(
      QKb, LFb, Wqkb, nullptr, nullptr, nullptr, TT, 2 * PP, HH);
  // 6) RoPE in place on bf16 QK
  rope_bf<<<dim3((TT * NHH * 2 * 32) / 256), blk, 0, stream>>>(QKb, rot);
  // 7) Causal flash attention -> AO (bf16)
  flash_attn<<<dim3(SS / 64, NHH, BBB), dim3(128), 0, stream>>>(AOb, QKb, Vb16);
  // 8) out = AO @ Wo^T                   (f32 final output)
  gemm_async<false, false, false><<<dim3(HH / 128, TT / 128), blk, 0, stream>>>(
      out, AOb, Wob, nullptr, nullptr, nullptr, TT, HH, PP);
}
